// Block_70849780515437
// MI455X (gfx1250) — compile-verified, matrix operand order checked by reading, not run on hardware
//
#include <hip/hip_runtime.h>
#include <stdint.h>

// ---------------------------------------------------------------------------
// Types for CDNA5 WMMA (wave32, 16x16x32 bf16 -> f32 accumulate)
// ---------------------------------------------------------------------------
typedef __attribute__((ext_vector_type(16))) __bf16 v16bf;
typedef __attribute__((ext_vector_type(8)))  float  v8f;

// ---------------------------------------------------------------------------
// Helpers
// ---------------------------------------------------------------------------
__device__ __forceinline__ uint16_t f2bf(float f) {
    uint32_t u = __float_as_uint(f);
    u += 0x7FFFu + ((u >> 16) & 1u);   // round-to-nearest-even
    return (uint16_t)(u >> 16);
}
__device__ __forceinline__ float bf2f(uint16_t v) {
    return __uint_as_float(((uint32_t)v) << 16);
}
__device__ __forceinline__ float sigf(float x) {
    return 1.0f / (1.0f + __expf(-x));
}
__device__ __forceinline__ float fast_tanh(float x) {
    float cx = fminf(fmaxf(x, -15.f), 15.f);
    float e = __expf(2.f * cx);
    return (e - 1.f) / (e + 1.f);
}

// K index inside a 16-bit A/B fragment: element i (0..15) of a lane whose
// half = lane>>4.  Matches ISA table "16-bit A-Matrix 16x32":
//   VGPR0..3 : K = {0..7} (lanes 0-15) / {8..15} (lanes 16-31)
//   VGPR4..7 : K = {16..23} / {24..31}
__device__ __forceinline__ int wmma_k_of(int i, int half) {
    return i + half * 8 + ((i >= 8) ? 8 : 0);
}

// Load an A fragment (16x32 bf16, row-major [16][kpad] in LDS, bf16 units).
// Pairs (K=2p,2p+1) are dword-contiguous -> 8 ds_load_b32 per fragment.
__device__ __forceinline__ v16bf load_afrag(const uint32_t* lds32, int kpad,
                                            int ktbase, int lane) {
    union { v16bf v; uint32_t u[8]; } fb;
    int half = lane >> 4, m = lane & 15;
    int rowbase = m * kpad + ktbase;
#pragma unroll
    for (int p = 0; p < 8; ++p) {
        int k = 2 * p + half * 8 + ((2 * p >= 8) ? 8 : 0);
        fb.u[p] = lds32[(rowbase + k) >> 1];
    }
    return fb.v;
}

// Load a pre-packed B fragment: 32 lanes x 32 contiguous bytes -> 2x b128.
__device__ __forceinline__ v16bf load_bfrag(const uint16_t* __restrict__ packed,
                                            int frag, int lane) {
    union { v16bf v; uint4 q[2]; } fb;
    const uint4* p = (const uint4*)(packed + ((size_t)frag * 512 + (size_t)lane * 16));
    fb.q[0] = p[0];
    fb.q[1] = p[1];
    return fb.v;
}

// ---------------------------------------------------------------------------
// One LSTM cell's gate GEMM:  gates[16][400] = X[16xKin]*Wih^T + H[16x100]*Whh^T + b
// XKT = number of 32-wide K tiles for the X operand (H operand is always 4).
// 8 waves split the 25 N tiles of the 400-wide gate dimension.
// ---------------------------------------------------------------------------
template <int XKT>
__device__ __forceinline__ void cell_gates(const uint32_t* xlds, int xkpad,
                                           const uint32_t* hlds,
                                           const uint16_t* __restrict__ wih,
                                           const uint16_t* __restrict__ whh,
                                           const float* __restrict__ bias,
                                           float* gates, int tid) {
    int lane = tid & 31, wave = tid >> 5;
    for (int nt = wave; nt < 25; nt += 8) {
        v8f acc = {};
#pragma unroll
        for (int kt = 0; kt < XKT; ++kt) {
            v16bf a = load_afrag(xlds, xkpad, kt * 32, lane);
            v16bf b = load_bfrag(wih, nt * XKT + kt, lane);
            acc = __builtin_amdgcn_wmma_f32_16x16x32_bf16(
                false, a, false, b, (short)0, acc, false, false);
        }
#pragma unroll
        for (int kt = 0; kt < 4; ++kt) {
            v16bf a = load_afrag(hlds, 128, kt * 32, lane);
            v16bf b = load_bfrag(whh, nt * 4 + kt, lane);
            acc = __builtin_amdgcn_wmma_f32_16x16x32_bf16(
                false, a, false, b, (short)0, acc, false, false);
        }
        int col = nt * 16 + (lane & 15);
        float bv = bias[col];
        int half = lane >> 4;
#pragma unroll
        for (int r = 0; r < 8; ++r)
            gates[(r + half * 8) * 400 + col] = acc[r] + bv;   // D layout: lane=col, vgpr r=row(+8*half)
    }
}

// Element-wise LSTM update (decoder variant; encoder inlines its own to also
// stream h to global).
__device__ __forceinline__ void cell_update(const float* gates, float* c,
                                            uint16_t* h16, int tid) {
    for (int e = tid; e < 1600; e += 256) {
        int r = e / 100, j = e - r * 100;
        float gi = gates[r * 400 + j];
        float gf = gates[r * 400 + 100 + j];
        float gg = gates[r * 400 + 200 + j];
        float go = gates[r * 400 + 300 + j];
        float cv = sigf(gf) * c[e] + sigf(gi) * fast_tanh(gg);
        c[e] = cv;
        h16[r * 128 + j] = f2bf(sigf(go) * fast_tanh(cv));
    }
}

// ---------------------------------------------------------------------------
// Weight packer: f32 [N=400][K] -> bf16 B-fragments (K padded to 32*ktiles).
// One block (32 lanes) per fragment; frag = nt*ktiles + kt.
// ---------------------------------------------------------------------------
__global__ void pack_b_kernel(const float* __restrict__ W,
                              uint16_t* __restrict__ out,
                              int N, int K, int ktiles) {
    int frag = blockIdx.x;
    int nt = frag / ktiles, kt = frag % ktiles;
    int l = threadIdx.x;
    int half = l >> 4;
    int ncol = nt * 16 + (l & 15);
    uint16_t* dst = out + (size_t)frag * 512 + (size_t)l * 16;
#pragma unroll
    for (int i = 0; i < 16; ++i) {
        int k = kt * 32 + wmma_k_of(i, half);
        float f = (ncol < N && k < K) ? W[(size_t)ncol * K + k] : 0.0f;
        dst[i] = f2bf(f);
    }
}

// inds / Cinds: per-batch flags from channel sums of x.
__global__ void inds_kernel(const float* __restrict__ x, int* __restrict__ inds,
                            int* __restrict__ cinds, int Bsz, int T, int NV) {
    int b = blockIdx.x * blockDim.x + threadIdx.x;
    if (b >= Bsz) return;
    float s1 = 0.f, s2 = 0.f;
    for (int t = 0; t < T; ++t) {
        s1 += x[((size_t)b * T + t) * NV + 1];
        s2 += x[((size_t)b * T + t) * NV + 2];
    }
    inds[b] = s1 > 0.f;
    cinds[b] = s2 > 0.f;
}

// ---------------------------------------------------------------------------
// Bidirectional LSTM layer: one block per (batch tile of 16, direction).
// Iterates all T steps internally; h/c live in LDS.
// ---------------------------------------------------------------------------
template <int XKT, int XKPAD>
__global__ __launch_bounds__(256) void lstm_layer_kernel(
    const float* __restrict__ xin, int xstride_b, int xstride_t, int kin,
    int zero12,
    const uint16_t* __restrict__ wih_all, const uint16_t* __restrict__ whh_all,
    const float* __restrict__ bias_all,
    float* __restrict__ yout, int T, int Bsz) {
    __shared__ uint32_t xlds[16 * XKPAD / 2];
    __shared__ uint32_t hlds[16 * 128 / 2];
    __shared__ float gates[16 * 400];
    __shared__ float cst[16 * 100];

    int tid = threadIdx.x;
    int btile = blockIdx.x, dir = blockIdx.y;
    const uint16_t* wih = wih_all + (size_t)dir * 25 * XKT * 512;
    const uint16_t* whh = whh_all + (size_t)dir * 25 * 4 * 512;
    const float* bias = bias_all + dir * 400;

    for (int e = tid; e < 16 * 128 / 2; e += 256) hlds[e] = 0u;
    for (int e = tid; e < 1600; e += 256) cst[e] = 0.f;
    __syncthreads();

    for (int s = 0; s < T; ++s) {
        int t = dir ? (T - 1 - s) : s;
        // stage x_t as bf16 [16][XKPAD] (zero pad + optional channel 1,2 mask)
        for (int e = tid; e < 16 * XKPAD; e += 256) {
            int r = e / XKPAD, k = e - r * XKPAD;
            int b = btile * 16 + r;
            float f = 0.f;
            if (b < Bsz && k < kin && !(zero12 && (k == 1 || k == 2)))
                f = xin[(size_t)b * xstride_b + (size_t)t * xstride_t + k];
            ((uint16_t*)xlds)[e] = f2bf(f);
        }
        __syncthreads();
        cell_gates<XKT>(xlds, XKPAD, hlds, wih, whh, bias, gates, tid);
        __syncthreads();
        for (int e = tid; e < 1600; e += 256) {
            int r = e / 100, j = e - r * 100;
            float gi = gates[r * 400 + j];
            float gf = gates[r * 400 + 100 + j];
            float gg = gates[r * 400 + 200 + j];
            float go = gates[r * 400 + 300 + j];
            float cv = sigf(gf) * cst[e] + sigf(gi) * fast_tanh(gg);
            cst[e] = cv;
            float h = sigf(go) * fast_tanh(cv);
            ((uint16_t*)hlds)[r * 128 + j] = f2bf(h);
            int b = btile * 16 + r;
            if (b < Bsz)
                yout[((size_t)b * T + t) * 200 + dir * 100 + j] = h;
        }
        __syncthreads();
    }
}

// ---------------------------------------------------------------------------
// Decoder: one block per batch tile owns all 12 cell states in LDS and runs
// the full 112-step autoregressive loop (3 stacks x 2 layers x 2 cells).
// ---------------------------------------------------------------------------
__global__ __launch_bounds__(256) void decoder_kernel(
    const float* __restrict__ y2,
    const uint16_t* __restrict__ wih0, const uint16_t* __restrict__ whh0,
    const float* __restrict__ b0,
    const uint16_t* __restrict__ wih1, const uint16_t* __restrict__ whh1,
    const float* __restrict__ b1,
    const float* __restrict__ lin_w, const float* __restrict__ lin_b,
    const int* __restrict__ inds, const int* __restrict__ cinds,
    float* __restrict__ out, int T, int F, int Bsz) {
    extern __shared__ uint8_t smem[];
    uint8_t* p = smem;
    uint32_t* lo32 = (uint32_t*)p;  p += 16 * 224 * 2;            // input bf16 [16][224]
    uint32_t* y032 = (uint32_t*)p;  p += 16 * 224 * 2;            // layer0 concat bf16
    uint16_t* h16b = (uint16_t*)p;  p += 12 * 16 * 128 * 2;       // 12x h bf16 [16][128]
    float*    cbas = (float*)p;     p += 12 * 16 * 100 * 4;       // 12x c f32 [16][100]
    float*    gates = (float*)p;    p += 16 * 400 * 4;            // gate buffer f32
    float*    lx = (float*)p;                                     // combine buffer f32 [16][200]
    uint16_t* lo16 = (uint16_t*)lo32;
    uint16_t* y016 = (uint16_t*)y032;

    int tid = threadIdx.x, btile = blockIdx.x;

    // init: lo = y2[:, T-1, :]; h = c = 0
    for (int e = tid; e < 16 * 224; e += 256) {
        int r = e / 224, k = e - r * 224;
        int b = btile * 16 + r;
        float f = (b < Bsz && k < 200) ? y2[((size_t)b * T + (T - 1)) * 200 + k] : 0.f;
        lo16[e] = f2bf(f);
    }
    for (int e = tid; e < 12 * 16 * 128; e += 256) h16b[e] = 0;
    for (int e = tid; e < 12 * 1600; e += 256) cbas[e] = 0.f;
    __syncthreads();

    for (int step = 0; step < F; ++step) {
        for (int stack = 0; stack < 3; ++stack) {
            // layer 0: two cells, both fed by lo
#pragma unroll
            for (int cell = 0; cell < 2; ++cell) {
                int ci = stack * 4 + cell, wi = stack * 2 + cell;
                cell_gates<7>(lo32, 224, (const uint32_t*)(h16b + ci * 2048),
                              wih0 + (size_t)wi * 25 * 7 * 512,
                              whh0 + (size_t)wi * 25 * 4 * 512,
                              b0 + wi * 400, gates, tid);
                __syncthreads();
                cell_update(gates, cbas + ci * 1600, h16b + ci * 2048, tid);
                __syncthreads();
            }
            // y0 = concat(h0, h1)
            for (int e = tid; e < 16 * 224; e += 256) {
                int r = e / 224, k = e - r * 224;
                uint16_t v = 0;
                if (k < 100)      v = h16b[(stack * 4 + 0) * 2048 + r * 128 + k];
                else if (k < 200) v = h16b[(stack * 4 + 1) * 2048 + r * 128 + (k - 100)];
                y016[e] = v;
            }
            __syncthreads();
            // layer 1: two cells fed by y0
#pragma unroll
            for (int cell = 0; cell < 2; ++cell) {
                int ci = stack * 4 + 2 + cell, wi = stack * 2 + cell;
                cell_gates<7>(y032, 224, (const uint32_t*)(h16b + ci * 2048),
                              wih1 + (size_t)wi * 25 * 7 * 512,
                              whh1 + (size_t)wi * 25 * 4 * 512,
                              b1 + wi * 400, gates, tid);
                __syncthreads();
                cell_update(gates, cbas + ci * 1600, h16b + ci * 2048, tid);
                __syncthreads();
            }
        }
        // combine: lx = yd - inds*relu(ys) + Cinds*relu(yc)
        for (int e = tid; e < 16 * 200; e += 256) {
            int r = e / 200, j = e - r * 200;
            int cs = (j >= 100) ? 3 : 2;
            int jj = (j >= 100) ? j - 100 : j;
            float yd = bf2f(h16b[(0 * 4 + cs) * 2048 + r * 128 + jj]);
            float ys = bf2f(h16b[(1 * 4 + cs) * 2048 + r * 128 + jj]);
            float yc = bf2f(h16b[(2 * 4 + cs) * 2048 + r * 128 + jj]);
            int b = btile * 16 + r;
            float v = yd;
            if (b < Bsz && inds[b])  v -= fmaxf(ys, 0.f);
            if (b < Bsz && cinds[b]) v += fmaxf(yc, 0.f);
            lx[e] = v;
        }
        __syncthreads();
        // next input + linear head
        for (int e = tid; e < 16 * 224; e += 256) {
            int r = e / 224, k = e - r * 224;
            lo16[e] = (k < 200) ? f2bf(lx[r * 200 + k]) : (uint16_t)0;
        }
        if (tid < 16) {
            float s = lin_b[0];
            for (int j = 0; j < 200; ++j) s += lx[tid * 200 + j] * lin_w[j];
            int b = btile * 16 + tid;
            if (b < Bsz) out[(size_t)b * F + step] = s;
        }
        __syncthreads();
    }
}

// ---------------------------------------------------------------------------
// Host-side launch
// ---------------------------------------------------------------------------
extern "C" void kernel_launch(void* const* d_in, const int* in_sizes, int n_in,
                              void* d_out, int out_size, void* d_ws, size_t ws_size,
                              hipStream_t stream) {
    (void)in_sizes; (void)n_in; (void)ws_size;
    const float* x        = (const float*)d_in[0];
    const float* enc_Wih0 = (const float*)d_in[2];
    const float* enc_Whh0 = (const float*)d_in[3];
    const float* enc_b0   = (const float*)d_in[4];
    const float* enc_Wih1 = (const float*)d_in[5];
    const float* enc_Whh1 = (const float*)d_in[6];
    const float* enc_b1   = (const float*)d_in[7];
    const float* dec_Wih0 = (const float*)d_in[8];
    const float* dec_Whh0 = (const float*)d_in[9];
    const float* dec_b0   = (const float*)d_in[10];
    const float* dec_Wih1 = (const float*)d_in[11];
    const float* dec_Whh1 = (const float*)d_in[12];
    const float* dec_b1   = (const float*)d_in[13];
    const float* lin_w    = (const float*)d_in[14];
    const float* lin_b    = (const float*)d_in[15];

    const int B = 500, T = 96, NV = 6;
    int F = out_size / B;   // forecast_length, derived deterministically

    uint8_t* ws = (uint8_t*)d_ws;
    size_t off = 0;
    auto carve = [&](size_t bytes) -> uint8_t* {
        uint8_t* q = ws + off;
        off += (bytes + 255) & ~(size_t)255;
        return q;
    };
    const size_t FRAG = 512 * sizeof(uint16_t);
    uint16_t* p_ewih0 = (uint16_t*)carve((size_t)2 * 25 * 1 * FRAG);
    uint16_t* p_ewhh0 = (uint16_t*)carve((size_t)2 * 25 * 4 * FRAG);
    uint16_t* p_ewih1 = (uint16_t*)carve((size_t)2 * 25 * 7 * FRAG);
    uint16_t* p_ewhh1 = (uint16_t*)carve((size_t)2 * 25 * 4 * FRAG);
    uint16_t* p_dwih0 = (uint16_t*)carve((size_t)6 * 25 * 7 * FRAG);
    uint16_t* p_dwhh0 = (uint16_t*)carve((size_t)6 * 25 * 4 * FRAG);
    uint16_t* p_dwih1 = (uint16_t*)carve((size_t)6 * 25 * 7 * FRAG);
    uint16_t* p_dwhh1 = (uint16_t*)carve((size_t)6 * 25 * 4 * FRAG);
    float* y1    = (float*)carve((size_t)B * T * 200 * 4);
    float* y2    = (float*)carve((size_t)B * T * 200 * 4);
    int* indsp   = (int*)carve((size_t)B * 4);
    int* cindsp  = (int*)carve((size_t)B * 4);

    auto pack = [&](const float* W, uint16_t* outp, int K, int nslices) {
        int ktiles = (K + 31) / 32;
        for (int s = 0; s < nslices; ++s)
            pack_b_kernel<<<dim3(25 * ktiles), dim3(32), 0, stream>>>(
                W + (size_t)s * 400 * K, outp + (size_t)s * 25 * ktiles * 512,
                400, K, ktiles);
    };
    pack(enc_Wih0, p_ewih0, 6, 2);
    pack(enc_Whh0, p_ewhh0, 100, 2);
    pack(enc_Wih1, p_ewih1, 200, 2);
    pack(enc_Whh1, p_ewhh1, 100, 2);
    pack(dec_Wih0, p_dwih0, 200, 6);
    pack(dec_Whh0, p_dwhh0, 100, 6);
    pack(dec_Wih1, p_dwih1, 200, 6);
    pack(dec_Whh1, p_dwhh1, 100, 6);

    inds_kernel<<<dim3((B + 127) / 128), dim3(128), 0, stream>>>(
        x, indsp, cindsp, B, T, NV);

    // encoder layer 0: K=6 (one 32-wide tile), channels 1,2 zeroed
    lstm_layer_kernel<1, 32><<<dim3(32, 2), dim3(256), 0, stream>>>(
        x, T * NV, NV, NV, 1, p_ewih0, p_ewhh0, enc_b0, y1, T, B);
    // encoder layer 1: K=200 (7 tiles)
    lstm_layer_kernel<7, 224><<<dim3(32, 2), dim3(256), 0, stream>>>(
        y1, T * 200, 200, 200, 0, p_ewih1, p_ewhh1, enc_b1, y2, T, B);

    const size_t SMEM = (size_t)(16 * 224 * 2) * 2 + (size_t)12 * 16 * 128 * 2 +
                        (size_t)12 * 16 * 100 * 4 + (size_t)16 * 400 * 4 +
                        (size_t)16 * 200 * 4;   // 178688 bytes, < 320KB WGP LDS
    (void)hipFuncSetAttribute((const void*)decoder_kernel,
                              hipFuncAttributeMaxDynamicSharedMemorySize, (int)SMEM);
    decoder_kernel<<<dim3((B + 15) / 16), dim3(256), SMEM, stream>>>(
        y2, p_dwih0, p_dwhh0, dec_b0, p_dwih1, p_dwhh1, dec_b1,
        lin_w, lin_b, indsp, cindsp, (float*)d_out, T, F, B);
}
